// RoMultiHeadAttention_31851477467659
// MI455X (gfx1250) — matrix operand order repaired and checked by reading, not compiled, not run on hardware
//
#include <hip/hip_runtime.h>
#include <hip/hip_bf16.h>

// ---------------------------------------------------------------------------
// Rotary MHA forward for MI455X (gfx1250, wave32).
//   - all GEMMs / attention einsums on v_wmma_f32_16x16x32_bf16 (f32 accum)
//   - one-shot fp32->bf16 conversion of activations + weights (pure-load GEMMs)
//   - RoPE sin/cos precomputed into a 512 KB table (no libm in hot loops)
//   - flash attention with K/V tiles double-buffered through LDS via
//     global_load_async_to_lds_b128 (ASYNCcnt) when the builtin is available
// Workspace: ~64.5 MiB of d_ws.
// ---------------------------------------------------------------------------

#define B_   2
#define S_   2048
#define D_   1024
#define H_   16
#define DH_  64
#define BS_  (B_ * S_)

typedef __attribute__((ext_vector_type(16))) __bf16        bf16x16;
typedef __attribute__((ext_vector_type(8)))  float         floatx8;
typedef __attribute__((ext_vector_type(8)))  unsigned int  uintx8;
typedef int v4i __attribute__((vector_size(16)));   // matches builtin param type

static_assert(sizeof(bf16x16) == 32, "bf16x16 size");
static_assert(sizeof(uintx8)  == 32, "uintx8 size");

#if __has_builtin(__builtin_amdgcn_global_load_async_to_lds_b128)
#define ASYNC_LDS 1
#else
#define ASYNC_LDS 0
#endif

#if ASYNC_LDS
#if __has_builtin(__builtin_amdgcn_s_wait_asynccnt)
#define ASYNC_WAIT(n) __builtin_amdgcn_s_wait_asynccnt(n)
#else
#define ASYNC_WAIT(n) asm volatile("s_wait_asynccnt %0" ::"i"(n) : "memory")
#endif
#else
#define ASYNC_WAIT(n) ((void)0)
#endif

// float -> bf16 RNE
static __device__ inline unsigned short f2bf(float f) {
    unsigned int u = __float_as_uint(f);
    u += 0x7FFFu + ((u >> 16) & 1u);
    return (unsigned short)(u >> 16);
}
static __device__ inline unsigned int pack2(float lo, float hi) {
    return (unsigned int)f2bf(lo) | ((unsigned int)f2bf(hi) << 16);
}

// D = A(16x32 bf16) * B(32x16 bf16) + C(16x16 f32)
static __device__ inline floatx8 wmma_bf16(uintx8 a, uintx8 b, floatx8 c) {
    return __builtin_amdgcn_wmma_f32_16x16x32_bf16(
        false, __builtin_bit_cast(bf16x16, a),
        false, __builtin_bit_cast(bf16x16, b),
        (short)0, c, false, false);
}

// A-fragment K index (CDNA5 ISA "16-bit A 16x32")
static __device__ inline int ka_map(int v, int half) {
    return (v < 4) ? (8 * half + 2 * v) : (16 + 8 * half + 2 * (v - 4));
}
// B-fragment K index (same K order as A fragment)
static __device__ inline int kb_map(int v, int half) {
    return (v < 4) ? (8 * half + 2 * v) : (16 + 8 * half + 2 * (v - 4));
}

// 16-byte global -> LDS stage (async DMA when available)
static __device__ inline void stage16(void* l, const void* g) {
#if ASYNC_LDS
    __builtin_amdgcn_global_load_async_to_lds_b128(
        (v4i*)g,                                   // flat/global source
        (__attribute__((address_space(3))) v4i*)(unsigned int)(unsigned long long)l,
        0, 0);
#else
    *(uint4*)l = *(const uint4*)g;
#endif
}

// ---------------------------------------------------------------------------
// Kernel 0a: fp32 -> packed bf16 (one pair per thread)
// ---------------------------------------------------------------------------
__global__ __launch_bounds__(256)
void cvt_bf16_kernel(const float* __restrict__ src, unsigned int* __restrict__ dst,
                     int npairs)
{
    const int i = blockIdx.x * blockDim.x + threadIdx.x;
    if (i < npairs) {
        const float2 f = ((const float2*)src)[i];
        dst[i] = pack2(f.x, f.y);
    }
}

// ---------------------------------------------------------------------------
// Kernel 0b: RoPE sin/cos table: tab[s*32 + i] = {sin(s*inv_i), cos(s*inv_i)}
// inv_i = exp(-(2i) * 4*ln(10)/DH) = exp(-i * ln(10)/8)
// ---------------------------------------------------------------------------
__global__ __launch_bounds__(256)
void rope_table_kernel(float2* __restrict__ tab)
{
    const int i = blockIdx.x * blockDim.x + threadIdx.x;
    if (i < S_ * 32) {
        const int s = i >> 5, fi = i & 31;
        const float inv = __expf(-(float)fi * 0.28782313662f);
        const float ang = (float)s * inv;
        tab[i] = make_float2(sinf(ang), cosf(ang));
    }
}

// ---------------------------------------------------------------------------
// Kernel 1: projection  out = x @ W^T + bias (+ optional RoPE) -> bf16 (B,H,S,DH)
// One wave computes a 16x64 output tile (A fragment reused across 4 WMMAs).
// ---------------------------------------------------------------------------
__global__ __launch_bounds__(32)
void proj_rope_kernel(const unsigned short* __restrict__ xb,   // (BS, D) bf16
                      const unsigned short* __restrict__ Wb,   // (D, D)  bf16
                      const float* __restrict__ bias,          // (D,)
                      const float2* __restrict__ rope,         // (S, 32)
                      unsigned short* __restrict__ out,        // (B,H,S,DH) bf16
                      int do_rope)
{
    const int lane = threadIdx.x;
    const int half = lane >> 4;
    const int ln   = lane & 15;
    const int tn   = blockIdx.x;        // 64-feature supertile
    const int tm   = blockIdx.y;        // 16-row tile

    const unsigned int* arow =
        (const unsigned int*)(xb + (size_t)(tm * 16 + ln) * D_);
    const unsigned int* wrow[4];
#pragma unroll
    for (int t = 0; t < 4; ++t)
        wrow[t] = (const unsigned int*)(Wb + (size_t)(tn * 64 + t * 16 + ln) * D_);

    __builtin_prefetch(arow, 0, 0);
    __builtin_prefetch(wrow[0], 0, 0);

    floatx8 c[4];
#pragma unroll
    for (int t = 0; t < 4; ++t) c[t] = (floatx8){0.f,0.f,0.f,0.f,0.f,0.f,0.f,0.f};

    for (int kk = 0; kk < D_; kk += 32) {
        uintx8 a;
#pragma unroll
        for (int v = 0; v < 8; ++v) a[v] = arow[(kk + ka_map(v, half)) >> 1];
#pragma unroll
        for (int t = 0; t < 4; ++t) {
            uintx8 b;
#pragma unroll
            for (int v = 0; v < 8; ++v) b[v] = wrow[t][(kk + kb_map(v, half)) >> 1];
            c[t] = wmma_bf16(a, b, c[t]);
        }
    }

#pragma unroll
    for (int t = 0; t < 4; ++t) {
        const int   col = tn * 64 + t * 16 + ln;   // C column = lane&15
        const float bv  = bias[col];
#pragma unroll
        for (int r = 0; r < 8; ++r) {
            const int row = tm * 16 + r + 8 * half;   // C row = r + 8*half
            const int bi  = row / S_;
            const int s   = row % S_;
            float val = c[t][r] + bv;

            if (do_rope) {
                // pair (2i, 2i+1) sits on adjacent lanes (N = lane&15)
                const float partner = __shfl_xor(val, 1, 32);
                const int   i  = (col & (DH_ - 1)) >> 1;
                const float2 sc = rope[s * 32 + i];
                if ((col & 1) == 0) val =  val * sc.x - partner * sc.y;  // x1*sin - x2*cos
                else                val = -partner * sc.y - val * sc.x;  // -x1*cos - x2*sin
            }

            const int h  = col >> 6;
            const int dh = col & (DH_ - 1);
            out[(((size_t)bi * H_ + h) * S_ + s) * DH_ + dh] = f2bf(val);
        }
    }
}

// ---------------------------------------------------------------------------
// Kernel 2: causal flash attention per (b,h). One wave per 16-row q-tile,
// 32-key tiles double-buffered through LDS with async DMA.
// ---------------------------------------------------------------------------
__global__ __launch_bounds__(32)
void flash_attn_kernel(const unsigned short* __restrict__ Q,  // (B,H,S,DH) bf16
                       const unsigned short* __restrict__ K,
                       const unsigned short* __restrict__ V,
                       unsigned short* __restrict__ O)        // (B,H,S,DH) bf16
{
    __shared__ __align__(16) unsigned short ldsK[2][32 * DH_]; // 2 x 4 KB
    __shared__ __align__(16) unsigned short ldsV[2][32 * DH_]; // 2 x 4 KB
    __shared__ __align__(16) unsigned short ldsP[16 * 32];     // 1 KB

    const int lane = threadIdx.x;
    const int half = lane >> 4;
    const int ln   = lane & 15;
    const int q0   = blockIdx.x * 16;
    const int bh   = blockIdx.y;

    const unsigned short* Qp = Q + (size_t)bh * S_ * DH_;
    const unsigned short* Kp = K + (size_t)bh * S_ * DH_;
    const unsigned short* Vp = V + (size_t)bh * S_ * DH_;

    // stage one 32-key tile of K and V (each 4 KB contiguous) into LDS buffer
    auto stage_tile = [&](int buf, int j0) {
        const char* ks = (const char*)(Kp + (size_t)j0 * DH_);
        const char* vs = (const char*)(Vp + (size_t)j0 * DH_);
        char* kd = (char*)ldsK[buf];
        char* vd = (char*)ldsV[buf];
#pragma unroll
        for (int c = 0; c < 8; ++c) {
            const int off = (c * 32 + lane) * 16;
            stage16(kd + off, ks + off);   // 8 issues x 512 B
            stage16(vd + off, vs + off);
        }
    };

    // Q A-fragments for dh = [0,32) and [32,64)
    uintx8 aq[2];
    {
        const unsigned int* qrow =
            (const unsigned int*)(Qp + (size_t)(q0 + ln) * DH_);
#pragma unroll
        for (int v = 0; v < 8; ++v) {
            const int ka = ka_map(v, half);
            aq[0][v] = qrow[(0  + ka) >> 1];
            aq[1][v] = qrow[(32 + ka) >> 1];
        }
    }

    float mrow[8], lrow[8];
    floatx8 acc[4];
#pragma unroll
    for (int r = 0; r < 8; ++r) { mrow[r] = -1e30f; lrow[r] = 0.f; }
#pragma unroll
    for (int t = 0; t < 4; ++t) acc[t] = (floatx8){0.f,0.f,0.f,0.f,0.f,0.f,0.f,0.f};

    const int jend = q0 + 16;
    int cur = 0;
    stage_tile(0, 0);

    for (int j0 = 0; j0 < jend; j0 += 32) {
        const bool have_next = (j0 + 32) < jend;
        if (have_next) stage_tile(cur ^ 1, j0 + 32);   // overlap DMA w/ compute
        if (have_next) { ASYNC_WAIT(16); } else { ASYNC_WAIT(0); }
        __syncthreads();

        const unsigned short* Kt = ldsK[cur];
        const unsigned short* Vt = ldsV[cur];

        // ---- scores: two 16-key C fragments, K-dim = DH (2 WMMA steps each)
        floatx8 sc[2];
#pragma unroll
        for (int kt = 0; kt < 2; ++kt) {
            const unsigned int* krow =
                (const unsigned int*)(Kt + (size_t)(16 * kt + ln) * DH_);
            floatx8 s = {0.f,0.f,0.f,0.f,0.f,0.f,0.f,0.f};
#pragma unroll
            for (int kk = 0; kk < 2; ++kk) {
                uintx8 bfrag;
#pragma unroll
                for (int v = 0; v < 8; ++v)
                    bfrag[v] = krow[(32 * kk + kb_map(v, half)) >> 1];
                s = wmma_bf16(aq[kk], bfrag, s);
            }
            sc[kt] = s;
        }

        // ---- causal mask + online softmax (per-row stats live at reg index r)
#pragma unroll
        for (int r = 0; r < 8; ++r) {
            const int qr = q0 + r + 8 * half;
            float v0 = ((j0 + ln)      > qr) ? -1e30f : sc[0][r];
            float v1 = ((j0 + 16 + ln) > qr) ? -1e30f : sc[1][r];

            float mx = fmaxf(v0, v1);
            mx = fmaxf(mx, __shfl_xor(mx, 1, 32));
            mx = fmaxf(mx, __shfl_xor(mx, 2, 32));
            mx = fmaxf(mx, __shfl_xor(mx, 4, 32));
            mx = fmaxf(mx, __shfl_xor(mx, 8, 32));
            const float mnew  = fmaxf(mrow[r], mx);
            const float scale = __expf(mrow[r] - mnew);
            mrow[r] = mnew;

            const float p0 = __expf(v0 - mnew);
            const float p1 = __expf(v1 - mnew);
            float ps = p0 + p1;
            ps += __shfl_xor(ps, 1, 32);
            ps += __shfl_xor(ps, 2, 32);
            ps += __shfl_xor(ps, 4, 32);
            ps += __shfl_xor(ps, 8, 32);
            lrow[r] = lrow[r] * scale + ps;

#pragma unroll
            for (int t = 0; t < 4; ++t) acc[t][r] *= scale;

            const int prow = r + 8 * half;
            ldsP[prow * 32 + ln]      = f2bf(p0);
            ldsP[prow * 32 + 16 + ln] = f2bf(p1);
        }
        __syncthreads();

        // ---- re-pack P (C layout in LDS) into A-fragment layout
        uintx8 ap;
        {
            const unsigned int* prow = (const unsigned int*)(ldsP + ln * 32);
#pragma unroll
            for (int v = 0; v < 8; ++v)
                ap[v] = prow[ka_map(v, half) >> 1];
        }

        // ---- P(16x32) @ V(32x64): 4 N-tiles of 16 over dh
#pragma unroll
        for (int t = 0; t < 4; ++t) {
            uintx8 bv;
#pragma unroll
            for (int v = 0; v < 8; ++v) {
                const int keyl = kb_map(v, half);
                const unsigned int lo = Vt[(size_t)keyl       * DH_ + 16 * t + ln];
                const unsigned int hi = Vt[(size_t)(keyl + 1) * DH_ + 16 * t + ln];
                bv[v] = lo | (hi << 16);
            }
            acc[t] = wmma_bf16(ap, bv, acc[t]);
        }
        __syncthreads();
        cur ^= 1;
    }

    // ---- epilogue: normalize and store bf16 (B,H,S,DH)
#pragma unroll
    for (int t = 0; t < 4; ++t) {
#pragma unroll
        for (int r = 0; r < 8; ++r) {
            const float o  = acc[t][r] / lrow[r];
            const int row = q0 + r + 8 * half;
            O[(size_t)bh * S_ * DH_ + (size_t)row * DH_ + 16 * t + ln] = f2bf(o);
        }
    }
}

// ---------------------------------------------------------------------------
// Kernel 3: final projection  out = Oattn @ Wo^T + bo  (fp32 output).
// ---------------------------------------------------------------------------
__global__ __launch_bounds__(32)
void out_proj_kernel(const unsigned short* __restrict__ Oin, // (B,H,S,DH) bf16
                     const unsigned short* __restrict__ Wob, // (D,D) bf16
                     const float* __restrict__ bo,           // (D,)
                     float* __restrict__ out)                // (B,S,D) fp32
{
    const int lane = threadIdx.x;
    const int half = lane >> 4;
    const int ln   = lane & 15;
    const int tn   = blockIdx.x;
    const int tm   = blockIdx.y;

    const int row_a = tm * 16 + ln;     // A row m = ln
    const int bi = row_a / S_;
    const int s  = row_a % S_;
    const unsigned short* abase = Oin + ((size_t)bi * H_) * S_ * DH_ + (size_t)s * DH_;
    const unsigned int* wrow[4];
#pragma unroll
    for (int t = 0; t < 4; ++t)
        wrow[t] = (const unsigned int*)(Wob + (size_t)(tn * 64 + t * 16 + ln) * D_);

    floatx8 c[4];
#pragma unroll
    for (int t = 0; t < 4; ++t) c[t] = (floatx8){0.f,0.f,0.f,0.f,0.f,0.f,0.f,0.f};

    for (int kk = 0; kk < D_; kk += 32) {
        const int h   = kk >> 6;        // 32-step stays inside one head
        const int dh0 = kk & (DH_ - 1);
        const unsigned int* arow =
            (const unsigned int*)(abase + (size_t)h * S_ * DH_ + dh0);
        uintx8 a;
#pragma unroll
        for (int v = 0; v < 8; ++v) a[v] = arow[ka_map(v, half) >> 1];
#pragma unroll
        for (int t = 0; t < 4; ++t) {
            uintx8 b;
#pragma unroll
            for (int v = 0; v < 8; ++v) b[v] = wrow[t][(kk + kb_map(v, half)) >> 1];
            c[t] = wmma_bf16(a, b, c[t]);
        }
    }

#pragma unroll
    for (int t = 0; t < 4; ++t) {
        const int   col = tn * 64 + t * 16 + ln;
        const float bv  = bo[col];
#pragma unroll
        for (int r = 0; r < 8; ++r) {
            const int row = tm * 16 + r + 8 * half;
            out[(size_t)row * D_ + col] = c[t][r] + bv;
        }
    }
}

// ---------------------------------------------------------------------------
extern "C" void kernel_launch(void* const* d_in, const int* in_sizes, int n_in,
                              void* d_out, int out_size, void* d_ws, size_t ws_size,
                              hipStream_t stream)
{
    (void)in_sizes; (void)n_in; (void)out_size; (void)ws_size;

    const float* q  = (const float*)d_in[0];
    const float* k  = (const float*)d_in[1];
    const float* v  = (const float*)d_in[2];
    const float* Wq = (const float*)d_in[3]; const float* bq = (const float*)d_in[4];
    const float* Wk = (const float*)d_in[5]; const float* bk = (const float*)d_in[6];
    const float* Wv = (const float*)d_in[7]; const float* bv = (const float*)d_in[8];
    const float* Wo = (const float*)d_in[9]; const float* bo = (const float*)d_in[10];
    // d_in[11] = causal mask (implemented analytically)
    float* out = (float*)d_out;

    // -------- workspace layout (bf16 elements unless noted) --------
    const size_t nE = (size_t)BS_ * D_;        // 4,194,304
    const size_t nW = (size_t)D_ * D_;         // 1,048,576
    unsigned short* p   = (unsigned short*)d_ws;
    unsigned short* qb  = p; p += nE;
    unsigned short* kb  = p; p += nE;
    unsigned short* vb  = p; p += nE;
    unsigned short* Wqb = p; p += nW;
    unsigned short* Wkb = p; p += nW;
    unsigned short* Wvb = p; p += nW;
    unsigned short* Wob = p; p += nW;
    unsigned short* Qr  = p; p += nE;
    unsigned short* Kr  = p; p += nE;
    unsigned short* Vr  = p; p += nE;
    unsigned short* Oa  = p; p += nE;
    float2* rope = (float2*)p;                 // S_*32 float2 (8-byte aligned)

    const int thr = 256;

    // fp32 -> bf16 conversions (one pass each)
    cvt_bf16_kernel<<<(int)(nE / 2 / thr), thr, 0, stream>>>(q,  (unsigned int*)qb,  (int)(nE / 2));
    cvt_bf16_kernel<<<(int)(nE / 2 / thr), thr, 0, stream>>>(k,  (unsigned int*)kb,  (int)(nE / 2));
    cvt_bf16_kernel<<<(int)(nE / 2 / thr), thr, 0, stream>>>(v,  (unsigned int*)vb,  (int)(nE / 2));
    cvt_bf16_kernel<<<(int)(nW / 2 / thr), thr, 0, stream>>>(Wq, (unsigned int*)Wqb, (int)(nW / 2));
    cvt_bf16_kernel<<<(int)(nW / 2 / thr), thr, 0, stream>>>(Wk, (unsigned int*)Wkb, (int)(nW / 2));
    cvt_bf16_kernel<<<(int)(nW / 2 / thr), thr, 0, stream>>>(Wv, (unsigned int*)Wvb, (int)(nW / 2));
    cvt_bf16_kernel<<<(int)(nW / 2 / thr), thr, 0, stream>>>(Wo, (unsigned int*)Wob, (int)(nW / 2));
    rope_table_kernel<<<(S_ * 32) / thr, thr, 0, stream>>>(rope);

    const dim3 gemmGrid(D_ / 64, BS_ / 16);   // (16, 256) 16x64 tiles
    const dim3 waveBlk(32);

    proj_rope_kernel<<<gemmGrid, waveBlk, 0, stream>>>(qb, Wqb, bq, rope, Qr, 1);
    proj_rope_kernel<<<gemmGrid, waveBlk, 0, stream>>>(kb, Wkb, bk, rope, Kr, 1);
    proj_rope_kernel<<<gemmGrid, waveBlk, 0, stream>>>(vb, Wvb, bv, rope, Vr, 0);

    flash_attn_kernel<<<dim3(S_ / 16, B_ * H_), waveBlk, 0, stream>>>(Qr, Kr, Vr, Oa);

    out_proj_kernel<<<gemmGrid, waveBlk, 0, stream>>>(Oa, Wob, bo, out);
}
